// STMGT_46806553592233
// MI455X (gfx1250) — matrix operand order repaired and checked by reading, not compiled, not run on hardware
//
#include <hip/hip_runtime.h>
#include <hip/hip_bf16.h>
#include <math.h>

// =====================================================================
// STMGT forward pass for MI455X (gfx1250, wave32).
// All dense GEMMs run through v_wmma_f32_16x16x32_f16 (f16 in, f32 acc).
// The weight panel for each workgroup (<=16KB) is staged into LDS once
// via global_load_async_to_lds_b128 (ASYNCcnt) and read as WMMA B
// fragments with ds_load_b128; activation rows stream directly from
// global (L2-resident) as 16B loads. Non-GEMM glue (layernorm, softmax
// attention, GAT segment softmax, transposes, gating, output heads) is
// plain VALU + atomics.
// =====================================================================

typedef __attribute__((ext_vector_type(16))) _Float16 v16h;
typedef __attribute__((ext_vector_type(8)))  _Float16 v8h;
typedef __attribute__((ext_vector_type(8)))  float    v8f;

namespace {

constexpr int Bc    = 2;
constexpr int Nc    = 325;
constexpr int Tc    = 48;
constexpr int Ec    = 2600;
constexpr int Dc    = 64;
constexpr int Hc    = 4;
constexpr int NBc   = 3;
constexpr int Kmix  = 3;
constexpr int PREDc = 12;
constexpr int TPc   = 12;
constexpr int HDc   = Dc / Hc;       // 16
constexpr int E2c   = Ec + Nc;       // 2925 edges incl. self loops
constexpr int BTc   = Bc * Tc;       // 96 graph instances
constexpr int Mc    = Bc * Nc * Tc;  // 31200 token rows
constexpr int BNc   = Bc * Nc;       // 650
constexpr int HD4   = Hc * Dc;       // 256 (GAT per-head dim is full D)
constexpr int KPMAX = 256;           // max padded K of any GEMM

__device__ __forceinline__ float gelu_exact(float x) {
  return 0.5f * x * (1.0f + erff(x * 0.70710678118654752f));
}

__device__ __forceinline__ void atomicMaxF(float* addr, float val) {
  int* ai  = (int*)addr;
  int  old = __float_as_int(*addr);
  while (__int_as_float(old) < val) {
    int assumed = old;
    old = atomicCAS(ai, assumed, __float_as_int(val));
    if (old == assumed) break;
  }
}

// ---------------------------------------------------------------------
// WMMA GEMM: C[M,N] = act(A[M,Kpad] * W^T[N,Kpad] + bias)
// A row-major f16 (K padded to mult of 32); BT is the weight
// pre-transposed to [n][k] f16. One wave computes a 16x16 tile; 8 waves
// per block tile a 64x32 macro tile. The 32-column B panel (32*Kpad
// halves, <=16KB) is DMA'd to LDS once per workgroup with
// global_load_async_to_lds_b128, then read as ISA-layout B fragments
// (two ds_load_b128 per K-step). A fragments are two global b128 loads.
// ---------------------------------------------------------------------
__global__ void k_gemm(const _Float16* __restrict__ A,
                       const _Float16* __restrict__ BT,
                       const float* __restrict__ bias,
                       float* __restrict__ C,
                       int M, int Kd, int Nn, int act) {
  // Single shared allocation -> LDS byte offset 0 for async-DMA VDST.
  __shared__ _Float16 sB[32 * KPMAX];

  const int lane = threadIdx.x & 31;
  const int wave = threadIdx.x >> 5;
  const int tn0  = blockIdx.y * 32;
  const int tm   = blockIdx.x * 64 + (wave & 3) * 16;
  const int tn   = tn0 + (wave >> 2) * 16;

  // ---- async-DMA the B panel (rows tn0..tn0+31 of BT) into LDS ----
  {
    const int chunks = (32 * Kd) >> 3;  // 8 halves (16B) per chunk
    for (int c = threadIdx.x; c < chunks; c += blockDim.x) {
      int nrow   = c / (Kd >> 3);
      int kchunk = (c - nrow * (Kd >> 3)) << 3;
      int gn = tn0 + nrow;
      if (gn >= Nn) gn = Nn - 1;
      unsigned loff = (unsigned)((nrow * Kd + kchunk) * 2);
      const _Float16* gp = BT + (size_t)gn * Kd + kchunk;
      asm volatile("global_load_async_to_lds_b128 %0, %1, off"
                   :: "v"(loff), "v"((unsigned long long)(uintptr_t)gp)
                   : "memory");
    }
    asm volatile("s_wait_asynccnt 0x0" ::: "memory");
    __syncthreads();
  }

  int m  = tm + (lane & 15);
  int am = m < M ? m : (M - 1);
  int n  = tn + (lane & 15);
  int rn = (n < Nn ? n : (Nn - 1)) - tn0;     // panel row in LDS, 0..31

  const int kbA = (lane >> 4) * 8;    // A: lane-half selects K sub-block of 8
  const int kbB = (lane >> 4) * 16;   // unused; B reads use kbA (K sub-block of 8)

  const _Float16* __restrict__ arow = A + (size_t)am * Kd;
  const _Float16* brow = sB + rn * Kd;

  v8f acc = {};
  for (int kc = 0; kc < Kd; kc += 32) {
    v8h a0 = *(const v8h*)(arow + kc + kbA);
    v8h a1 = *(const v8h*)(arow + kc + 16 + kbA);
    v8h b0 = *(const v8h*)(brow + kc + kbA);
    v8h b1 = *(const v8h*)(brow + kc + 16 + kbA);
    if (kc + 32 < Kd) __builtin_prefetch(arow + kc + 32, 0, 0);
    v16h a, b;
#pragma unroll
    for (int i = 0; i < 8; ++i) {
      a[i] = a0[i]; a[8 + i] = a1[i];
      b[i] = b0[i]; b[8 + i] = b1[i];
    }
    acc = __builtin_amdgcn_wmma_f32_16x16x32_f16(
        false, a, false, b, (short)0, acc, false, false);
  }

  // C/D layout: VGPR r -> M = r (+8 for upper lane half), N = lane & 15.
  const int row0 = tm + ((lane >> 4) ? 8 : 0);
  const int cn   = tn + (lane & 15);
  if (cn < Nn) {
    const float bv = bias ? bias[cn] : 0.0f;
#pragma unroll
    for (int r = 0; r < 8; ++r) {
      const int rm = row0 + r;
      if (rm < M) {
        float v = acc[r] + bv;
        if (act == 1)      v = gelu_exact(v);
        else if (act == 2) v = 1.0f / (1.0f + __expf(-v));
        C[(size_t)rm * Nn + cn] = v;
      }
    }
  }
}

// Weight convert: W[K0,N0] f32 -> WT[N0][Kpad] f16, zero-padded K.
__global__ void k_wt2h(const float* __restrict__ w, _Float16* __restrict__ wt,
                       int K0, int N0, int Kpad) {
  int idx = blockIdx.x * blockDim.x + threadIdx.x;
  if (idx >= N0 * Kpad) return;
  int n = idx / Kpad, k = idx % Kpad;
  wt[idx] = (k < K0) ? (_Float16)w[(size_t)k * N0 + n] : (_Float16)0.0f;
}

__global__ void k_f2h(const float* __restrict__ x, _Float16* __restrict__ y,
                      size_t n) {
  size_t i = (size_t)blockIdx.x * blockDim.x + threadIdx.x;
  if (i < n) y[i] = (_Float16)x[i];
}

// Strided convert for the gate concat: src [rows,cols] -> dst[r*ld+off+c].
__global__ void k_f2h_str(const float* __restrict__ x, _Float16* __restrict__ y,
                          int rows, int cols, int ld, int off) {
  size_t i = (size_t)blockIdx.x * blockDim.x + threadIdx.x;
  if (i >= (size_t)rows * cols) return;
  int r = (int)(i / cols), c = (int)(i % cols);
  y[(size_t)r * ld + off + c] = (_Float16)x[i];
}

__global__ void k_fill(float* __restrict__ p, float v, size_t n) {
  size_t i = (size_t)blockIdx.x * blockDim.x + threadIdx.x;
  if (i < n) p[i] = v;
}

// ------------------------- embedding stage ---------------------------
__global__ void k_embed_traffic(const float* __restrict__ xtr,
                                const float* __restrict__ Wtr,
                                const float* __restrict__ btr,
                                float* __restrict__ x) {
  size_t i = (size_t)blockIdx.x * blockDim.x + threadIdx.x;
  if (i >= (size_t)Mc * Dc) return;
  int d = (int)(i & 63);
  size_t bnt = i >> 6;
  float xn = (xtr[bnt] - 18.72f) / (7.03f + 1e-8f);
  x[i] = xn * Wtr[d] + btr[d];
}

// Build padded temporal-embedding input: rows B*T, 64 cols (50 used).
__global__ void k_build_temb(const int* __restrict__ hour,
                             const int* __restrict__ dow,
                             const int* __restrict__ wk,
                             const float* __restrict__ Edow,
                             const float* __restrict__ Ewk,
                             float* __restrict__ tin) {
  int i = blockIdx.x * blockDim.x + threadIdx.x;
  if (i >= BTc * 64) return;
  int row = i >> 6, c = i & 63;
  float v = 0.0f;
  float hr = (float)hour[row] * (2.0f * 3.14159265358979323846f / 24.0f);
  if (c == 0)       v = sinf(hr);
  else if (c == 1)  v = cosf(hr);
  else if (c < 34)  v = Edow[dow[row] * 32 + (c - 2)];
  else if (c < 50)  v = Ewk[wk[row] * 16 + (c - 34)];
  tin[i] = v;
}

__global__ void k_add_temb(float* __restrict__ x,
                           const float* __restrict__ temb) {
  size_t i = (size_t)blockIdx.x * blockDim.x + threadIdx.x;
  if (i >= (size_t)Mc * Dc) return;
  int d = (int)(i & 63);
  size_t r = i >> 6;                 // (b*N+n)*T+t
  int t = (int)(r % Tc);
  int b = (int)(r / ((size_t)Nc * Tc));
  x[i] += temb[((size_t)(b * Tc + t) << 6) + d];
}

// ------------------------- layout transposes -------------------------
__global__ void k_bntd2btnd(const float* __restrict__ x, float* __restrict__ y) {
  size_t i = (size_t)blockIdx.x * blockDim.x + threadIdx.x;
  if (i >= (size_t)Mc * Dc) return;
  int d = (int)(i & 63);
  size_t r = i >> 6;                 // (b*N+n)*T+t
  int t = (int)(r % Tc);
  size_t bn = r / Tc;
  int n = (int)(bn % Nc), b = (int)(bn / Nc);
  y[((((size_t)b * Tc + t) * Nc + n) << 6) + d] = x[i];
}

__global__ void k_btnd2bntd(const float* __restrict__ x, float* __restrict__ y) {
  size_t i = (size_t)blockIdx.x * blockDim.x + threadIdx.x;
  if (i >= (size_t)Mc * Dc) return;
  int d = (int)(i & 63);
  size_t r = i >> 6;                 // (b*T+t)*N+n
  int n = (int)(r % Nc);
  size_t bt = r / Nc;
  int t = (int)(bt % Tc), b = (int)(bt / Tc);
  y[((((size_t)b * Nc + n) * Tc + t) << 6) + d] = x[i];
}

// --------------------------- layernorm -------------------------------
// y = LN(x + r?) * g + b, row length 64, one wave32 per row.
__global__ void k_ln(const float* __restrict__ x, const float* __restrict__ r,
                     const float* __restrict__ g, const float* __restrict__ bb,
                     float* __restrict__ y, int rows) {
  int row  = blockIdx.x * (blockDim.x >> 5) + (threadIdx.x >> 5);
  int lane = threadIdx.x & 31;
  if (row >= rows) return;
  size_t base = (size_t)row << 6;
  float v0 = x[base + lane], v1 = x[base + lane + 32];
  if (r) { v0 += r[base + lane]; v1 += r[base + lane + 32]; }
  float s = v0 + v1;
#pragma unroll
  for (int o = 16; o > 0; o >>= 1) s += __shfl_xor(s, o);
  float mean = s * (1.0f / 64.0f);
  float d0 = v0 - mean, d1 = v1 - mean;
  float vs = d0 * d0 + d1 * d1;
#pragma unroll
  for (int o = 16; o > 0; o >>= 1) vs += __shfl_xor(vs, o);
  float inv = rsqrtf(vs * (1.0f / 64.0f) + 1e-5f);
  y[base + lane]      = d0 * inv * g[lane] + bb[lane];
  y[base + lane + 32] = d1 * inv * g[lane + 32] + bb[lane + 32];
}

// ---------------------- temporal self-attention ----------------------
// Q,K,V: [B*N*T, 64]; one block per (b*N+n, h); thread q handles a row.
__global__ void k_attn_time(const float* __restrict__ Q,
                            const float* __restrict__ Km,
                            const float* __restrict__ V,
                            float* __restrict__ O) {
  int bn = blockIdx.x / Hc;
  int h  = blockIdx.x % Hc;
  int q  = threadIdx.x;
  if (q >= Tc) return;
  const float scale = 0.25f;  // 1/sqrt(16)
  size_t base = (size_t)bn * Tc * Dc + h * HDc;
  float qv[HDc];
#pragma unroll
  for (int d = 0; d < HDc; ++d) qv[d] = Q[base + (size_t)q * Dc + d];
  float s[Tc];
  float mx = -1e30f;
  for (int k = 0; k < Tc; ++k) {
    const float* kp = Km + base + (size_t)k * Dc;
    float dot = 0.0f;
#pragma unroll
    for (int d = 0; d < HDc; ++d) dot += qv[d] * kp[d];
    dot *= scale;
    s[k] = dot;
    mx = fmaxf(mx, dot);
  }
  float den = 0.0f;
  for (int k = 0; k < Tc; ++k) { s[k] = __expf(s[k] - mx); den += s[k]; }
  float invd = 1.0f / den;
  float o[HDc] = {};
  for (int k = 0; k < Tc; ++k) {
    float a = s[k] * invd;
    const float* vp = V + base + (size_t)k * Dc;
#pragma unroll
    for (int d = 0; d < HDc; ++d) o[d] += a * vp[d];
  }
#pragma unroll
  for (int d = 0; d < HDc; ++d) O[base + (size_t)q * Dc + d] = o[d];
}

// ------------------------------ GATv2 --------------------------------
// xl,xr: [BT, N, 256]; e/ex: [BT, E2, H]; mx/den: [BT, N, H]
__global__ void k_gat_edge(const float* __restrict__ xl,
                           const float* __restrict__ xr,
                           const float* __restrict__ att,
                           const int* __restrict__ src,
                           const int* __restrict__ dst,
                           float* __restrict__ e) {
  int i = blockIdx.x * blockDim.x + threadIdx.x;
  if (i >= BTc * E2c * Hc) return;
  int h = i % Hc, t = i / Hc;
  int eg = t % E2c, m = t / E2c;
  int s = src[eg], dn = dst[eg];
  const float* pl = xl + ((size_t)m * Nc + s)  * HD4 + h * Dc;
  const float* pr = xr + ((size_t)m * Nc + dn) * HD4 + h * Dc;
  const float* pa = att + h * Dc;
  float acc = 0.0f;
#pragma unroll 8
  for (int d = 0; d < Dc; ++d) {
    float v = pl[d] + pr[d];
    v = v > 0.0f ? v : 0.2f * v;
    acc += v * pa[d];
  }
  e[i] = acc;
}

__global__ void k_seg_max(const float* __restrict__ e,
                          const int* __restrict__ dst,
                          float* __restrict__ mx) {
  int i = blockIdx.x * blockDim.x + threadIdx.x;
  if (i >= BTc * E2c * Hc) return;
  int h = i % Hc, t = i / Hc;
  int eg = t % E2c, m = t / E2c;
  atomicMaxF(&mx[((size_t)m * Nc + dst[eg]) * Hc + h], e[i]);
}

__global__ void k_seg_exp(float* __restrict__ e,
                          const int* __restrict__ dst,
                          const float* __restrict__ mx,
                          float* __restrict__ den) {
  int i = blockIdx.x * blockDim.x + threadIdx.x;
  if (i >= BTc * E2c * Hc) return;
  int h = i % Hc, t = i / Hc;
  int eg = t % E2c, m = t / E2c;
  float ex = __expf(e[i] - mx[((size_t)m * Nc + dst[eg]) * Hc + h]);
  e[i] = ex;
  atomicAdd(&den[((size_t)m * Nc + dst[eg]) * Hc + h], ex);
}

__global__ void k_gat_agg(const float* __restrict__ ex,
                          const float* __restrict__ den,
                          const float* __restrict__ xl,
                          const int* __restrict__ src,
                          const int* __restrict__ dst,
                          float* __restrict__ gacc) {
  int i = blockIdx.x * blockDim.x + threadIdx.x;
  if (i >= BTc * E2c * Hc) return;
  int h = i % Hc, t = i / Hc;
  int eg = t % E2c, m = t / E2c;
  int s = src[eg], dn = dst[eg];
  float a = ex[i] / den[((size_t)m * Nc + dn) * Hc + h];
  const float* pl = xl + ((size_t)m * Nc + s) * HD4 + h * Dc;
  float* po = gacc + ((size_t)m * Nc + dn) * HD4 + h * Dc;
#pragma unroll 4
  for (int d = 0; d < Dc; ++d) atomicAdd(po + d, a * pl[d]);
}

__global__ void k_gat_fin(const float* __restrict__ gacc,
                          const float* __restrict__ bias,
                          float* __restrict__ y) {
  size_t i = (size_t)blockIdx.x * blockDim.x + threadIdx.x;
  if (i >= (size_t)Mc * Dc) return;
  int d = (int)(i & 63);
  size_t mn = i >> 6;
  const float* p = gacc + mn * HD4 + d;
  y[i] = (p[0] + p[64] + p[128] + p[192]) * 0.25f + bias[d];
}

// ------------------------------ gating -------------------------------
__global__ void k_gate(const float* __restrict__ g, const float* __restrict__ xs,
                       const float* __restrict__ xt, const float* __restrict__ x,
                       float* __restrict__ y, size_t n) {
  size_t i = (size_t)blockIdx.x * blockDim.x + threadIdx.x;
  if (i < n) {
    float gg = g[i];
    y[i] = gg * xs[i] + (1.0f - gg) * xt[i] + x[i];
  }
}

__global__ void k_meanT(const float* __restrict__ x, float* __restrict__ y) {
  int i = blockIdx.x * blockDim.x + threadIdx.x;
  if (i >= BNc * Dc) return;
  int d = i & 63, bn = i >> 6;
  float s = 0.0f;
  for (int t = 0; t < Tc; ++t) s += x[(((size_t)bn * Tc + t) << 6) + d];
  y[i] = s * (1.0f / Tc);
}

// weather: normalize + 3->64 projection (K too small for WMMA)
__global__ void k_weather(const float* __restrict__ xw,
                          const float* __restrict__ Wwp,
                          const float* __restrict__ bwp,
                          float* __restrict__ wp) {
  int i = blockIdx.x * blockDim.x + threadIdx.x;
  if (i >= Bc * TPc * Dc) return;
  int d = i & 63, r = i >> 6;
  const float wm[3] = {25.0f, 15.0f, 1.0f};
  const float ws[3] = {5.0f, 10.0f, 2.0f};
  float acc = bwp[d];
#pragma unroll
  for (int c = 0; c < 3; ++c) {
    float v = (xw[r * 3 + c] - wm[c]) / (ws[c] + 1e-8f);
    acc += v * Wwp[c * Dc + d];
  }
  wp[i] = acc;
}

// cross attention: Q [B*N,64], K/V [B*TP,64]; one thread per (b,n,h).
__global__ void k_attn_cross(const float* __restrict__ Q,
                             const float* __restrict__ Kc,
                             const float* __restrict__ Vc,
                             float* __restrict__ O) {
  int i = blockIdx.x * blockDim.x + threadIdx.x;
  if (i >= BNc * Hc) return;
  int h = i % Hc, bn = i / Hc, b = bn / Nc;
  const float* q = Q + (size_t)bn * Dc + h * HDc;
  float s[TPc];
  float mx = -1e30f;
  for (int k = 0; k < TPc; ++k) {
    const float* kp = Kc + ((size_t)b * TPc + k) * Dc + h * HDc;
    float dot = 0.0f;
#pragma unroll
    for (int d = 0; d < HDc; ++d) dot += q[d] * kp[d];
    dot *= 0.25f;
    s[k] = dot;
    mx = fmaxf(mx, dot);
  }
  float den = 0.0f;
  for (int k = 0; k < TPc; ++k) { s[k] = __expf(s[k] - mx); den += s[k]; }
  float o[HDc] = {};
  for (int k = 0; k < TPc; ++k) {
    float a = s[k] / den;
    const float* vp = Vc + ((size_t)b * TPc + k) * Dc + h * HDc;
#pragma unroll
    for (int d = 0; d < HDc; ++d) o[d] += a * vp[d];
  }
  float* op = O + (size_t)bn * Dc + h * HDc;
#pragma unroll
  for (int d = 0; d < HDc; ++d) op[d] = o[d];
}

// output heads: mu | clip(exp(sg),0.1,10) | logit, concatenated flat.
__global__ void k_heads(const float* __restrict__ mu,
                        const float* __restrict__ sg,
                        const float* __restrict__ pi,
                        float* __restrict__ out) {
  int i = blockIdx.x * blockDim.x + threadIdx.x;
  const int n = BNc * PREDc * Kmix;  // 23400
  if (i >= n) return;
  out[i] = mu[i];
  float e = __expf(sg[i]);
  out[n + i] = fminf(fmaxf(e, 0.1f), 10.0f);
  out[2 * n + i] = pi[i];
}

}  // namespace

// =====================================================================
// Host orchestration
// =====================================================================
extern "C" void kernel_launch(void* const* d_in, const int* in_sizes, int n_in,
                              void* d_out, int out_size, void* d_ws, size_t ws_size,
                              hipStream_t stream) {
  (void)in_sizes; (void)n_in; (void)out_size; (void)ws_size;

  // ---- input pointers, pytree leaf order (dicts sorted alphabetically) ----
  int pi_ = 0;
  auto nf = [&]() { return (const float*)d_in[pi_++]; };
  const float* x_traffic = nf();
  const float* x_weather = nf();

  const float *Edow = nf(), *Ewk = nf();
  const float *Wmu = nf(), *Wpi = nf(), *Wsg = nf(), *Wte = nf();
  const float *Wtr = nf(), *Wwp = nf();

  struct BlkP {
    const float *W1, *W2, *Wg;
    const float *aWk, *aWo, *aWq, *aWv, *abk, *abo, *abq, *abv;
    const float *b1, *b2, *bg;
    const float *gWl, *gWr, *gatt, *gbias, *gbl, *gbr;
    const float *lnf_g, *lnf_b, *lns_g, *lns_b;
    const float *lnt1_g, *lnt1_b, *lnt2_g, *lnt2_b;
  } blk[NBc];
  for (int b = 0; b < NBc; ++b) {
    BlkP& p = blk[b];
    p.W1 = nf(); p.W2 = nf(); p.Wg = nf();
    p.aWk = nf(); p.aWo = nf(); p.aWq = nf(); p.aWv = nf();
    p.abk = nf(); p.abo = nf(); p.abq = nf(); p.abv = nf();
    p.b1 = nf(); p.b2 = nf(); p.bg = nf();
    p.gWl = nf(); p.gWr = nf(); p.gatt = nf(); p.gbias = nf();
    p.gbl = nf(); p.gbr = nf();
    p.lnf_g = nf(); p.lnf_b = nf();
    p.lns_g = nf(); p.lns_b = nf();
    p.lnt1_g = nf(); p.lnt1_b = nf();
    p.lnt2_g = nf(); p.lnt2_b = nf();
  }
  const float *bmu = nf(), *bpi = nf(), *bsg = nf(), *bte = nf();
  const float *btr = nf(), *bwp = nf();
  const float *cWk = nf(), *cWo = nf(), *cWq = nf(), *cWv = nf();
  const float *cbk = nf(), *cbo = nf(), *cbq = nf(), *cbv = nf();
  const float *lnc_g = nf(), *lnc_b = nf();
  const int* edge_index = (const int*)d_in[pi_++];
  const int* hour = (const int*)d_in[pi_++];
  const int* dow  = (const int*)d_in[pi_++];
  const int* wknd = (const int*)d_in[pi_++];
  const int* srcE = edge_index;
  const int* dstE = edge_index + E2c;

  // ---- workspace bump allocator ----
  char* wsp = (char*)d_ws;
  auto bump = [&](size_t bytes) {
    void* r = wsp;
    wsp += (bytes + 255) & ~(size_t)255;
    return r;
  };
  const size_t MD  = (size_t)Mc * Dc;       // 31200*64
  const size_t MD4 = (size_t)Mc * HD4;      // 31200*256
  float* bufX  = (float*)bump(MD * 4);
  float* bufXS = (float*)bump(MD * 4);
  float* bufXT = (float*)bump(MD * 4);
  float* bufS  = (float*)bump(MD * 4);
  float* t1    = (float*)bump(MD * 4);
  float* t2    = (float*)bump(MD * 4);
  float* t3    = (float*)bump(MD * 4);
  float* R1    = (float*)bump(MD4 * 4);     // xl
  float* R2    = (float*)bump(MD4 * 4);     // xr / ffn h1
  float* R3    = (float*)bump(MD4 * 4);     // GAT accumulator
  _Float16* hA = (_Float16*)bump(MD4 * 2);  // activation halves (<= 256 cols)
  _Float16* hW = (_Float16*)bump(64 * 1024);// weight halves (<= 256x64)
  float* eBuf  = (float*)bump((size_t)BTc * E2c * Hc * 4);
  float* mxB   = (float*)bump((size_t)BTc * Nc * Hc * 4);
  float* denB  = (float*)bump((size_t)BTc * Nc * Hc * 4);
  float* tin   = (float*)bump((size_t)BTc * Dc * 4);
  float* temb  = (float*)bump((size_t)BTc * Dc * 4);
  float* wp    = (float*)bump((size_t)Bc * TPc * Dc * 4);
  float* xm    = (float*)bump((size_t)BNc * Dc * 4);
  float* c1    = (float*)bump((size_t)BNc * Dc * 4);
  float* c2    = (float*)bump((size_t)BNc * Dc * 4);
  float* c3    = (float*)bump((size_t)BNc * Dc * 4);
  float* c4    = (float*)bump((size_t)BNc * Dc * 4);
  float* c5    = (float*)bump((size_t)BNc * Dc * 4);
  float* xf    = (float*)bump((size_t)BNc * Dc * 4);
  float* muR   = (float*)bump((size_t)BNc * PREDc * Kmix * 4);
  float* sgR   = (float*)bump((size_t)BNc * PREDc * Kmix * 4);
  float* piR   = (float*)bump((size_t)BNc * PREDc * Kmix * 4);
  _Float16* hKV = (_Float16*)bump((size_t)Bc * TPc * Dc * 2);

  auto gblk = [](size_t n) { return dim3((unsigned)((n + 255) / 256)); };

  // WMMA GEMM wrapper: convert weight (transpose+pad) then launch.
  auto gemm = [&](const _Float16* A, const float* W, int K0, int N0, int Kpad,
                  const float* bias, float* Cout, int Mrows, int act) {
    k_wt2h<<<gblk((size_t)N0 * Kpad), 256, 0, stream>>>(W, hW, K0, N0, Kpad);
    dim3 g((Mrows + 63) / 64, (N0 + 31) / 32);
    k_gemm<<<g, 256, 0, stream>>>(A, hW, bias, Cout, Mrows, Kpad, N0, act);
  };
  auto f2h = [&](const float* x, _Float16* y, size_t n) {
    k_f2h<<<gblk(n), 256, 0, stream>>>(x, y, n);
  };
  auto ln = [&](const float* x, const float* r, const float* g, const float* b,
                float* y, int rows) {
    k_ln<<<dim3((rows + 7) / 8), 256, 0, stream>>>(x, r, g, b, y, rows);
  };

  // ================= embedding =================
  k_embed_traffic<<<gblk(MD), 256, 0, stream>>>(x_traffic, Wtr, btr, bufX);
  k_build_temb<<<gblk((size_t)BTc * 64), 256, 0, stream>>>(hour, dow, wknd,
                                                           Edow, Ewk, tin);
  f2h(tin, hA, (size_t)BTc * 64);
  gemm(hA, Wte, 50, Dc, 64, bte, temb, BTc, 0);
  k_add_temb<<<gblk(MD), 256, 0, stream>>>(bufX, temb);

  // ================= blocks =================
  const int nEH = BTc * E2c * Hc;
  for (int b = 0; b < NBc; ++b) {
    const BlkP& p = blk[b];

    // ---- spatial path (GATv2 on (B*T) graphs) ----
    k_bntd2btnd<<<gblk(MD), 256, 0, stream>>>(bufX, bufS);
    f2h(bufS, hA, MD);
    gemm(hA, p.gWl, Dc, HD4, Dc, p.gbl, R1, Mc, 0);   // xl
    gemm(hA, p.gWr, Dc, HD4, Dc, p.gbr, R2, Mc, 0);   // xr
    k_fill<<<gblk((size_t)BTc * Nc * Hc), 256, 0, stream>>>(mxB, -1e30f,
                                                  (size_t)BTc * Nc * Hc);
    k_fill<<<gblk((size_t)BTc * Nc * Hc), 256, 0, stream>>>(denB, 0.0f,
                                                  (size_t)BTc * Nc * Hc);
    k_fill<<<gblk(MD4), 256, 0, stream>>>(R3, 0.0f, MD4);
    k_gat_edge<<<gblk(nEH), 256, 0, stream>>>(R1, R2, p.gatt, srcE, dstE, eBuf);
    k_seg_max<<<gblk(nEH), 256, 0, stream>>>(eBuf, dstE, mxB);
    k_seg_exp<<<gblk(nEH), 256, 0, stream>>>(eBuf, dstE, mxB, denB);
    k_gat_agg<<<gblk(nEH), 256, 0, stream>>>(eBuf, denB, R1, srcE, dstE, R3);
    k_gat_fin<<<gblk(MD), 256, 0, stream>>>(R3, p.gbias, bufS);
    k_btnd2bntd<<<gblk(MD), 256, 0, stream>>>(bufS, bufXS);
    ln(bufXS, nullptr, p.lns_g, p.lns_b, bufXS, Mc);

    // ---- temporal path (MHA over T) ----
    f2h(bufX, hA, MD);
    gemm(hA, p.aWq, Dc, Dc, Dc, p.abq, t1, Mc, 0);
    gemm(hA, p.aWk, Dc, Dc, Dc, p.abk, t2, Mc, 0);
    gemm(hA, p.aWv, Dc, Dc, Dc, p.abv, t3, Mc, 0);
    k_attn_time<<<dim3(BNc * Hc), 64, 0, stream>>>(t1, t2, t3, bufS);
    f2h(bufS, hA, MD);
    gemm(hA, p.aWo, Dc, Dc, Dc, p.abo, t1, Mc, 0);
    ln(t1, bufX, p.lnt1_g, p.lnt1_b, bufXT, Mc);

    // ---- FFN ----
    f2h(bufXT, hA, MD);
    gemm(hA, p.W1, Dc, 4 * Dc, Dc, p.b1, R2, Mc, 1);       // gelu fused
    f2h(R2, hA, MD4);
    gemm(hA, p.W2, 4 * Dc, Dc, 4 * Dc, p.b2, t1, Mc, 0);
    ln(t1, bufXT, p.lnt2_g, p.lnt2_b, bufXT, Mc);

    // ---- gated fusion ----
    k_f2h_str<<<gblk(MD), 256, 0, stream>>>(bufXS, hA, Mc, Dc, 2 * Dc, 0);
    k_f2h_str<<<gblk(MD), 256, 0, stream>>>(bufXT, hA, Mc, Dc, 2 * Dc, Dc);
    gemm(hA, p.Wg, 2 * Dc, Dc, 2 * Dc, p.bg, t2, Mc, 2);   // sigmoid fused
    k_gate<<<gblk(MD), 256, 0, stream>>>(t2, bufXS, bufXT, bufX, t3, MD);
    ln(t3, nullptr, p.lnf_g, p.lnf_b, bufX, Mc);
  }

  // ================= pooling + cross attention =================
  k_meanT<<<gblk((size_t)BNc * Dc), 256, 0, stream>>>(bufX, xm);
  k_weather<<<gblk((size_t)Bc * TPc * Dc), 256, 0, stream>>>(x_weather, Wwp,
                                                             bwp, wp);
  f2h(xm, hA, (size_t)BNc * Dc);
  gemm(hA, cWq, Dc, Dc, Dc, cbq, c1, BNc, 0);
  f2h(wp, hKV, (size_t)Bc * TPc * Dc);
  gemm(hKV, cWk, Dc, Dc, Dc, cbk, c2, Bc * TPc, 0);
  gemm(hKV, cWv, Dc, Dc, Dc, cbv, c3, Bc * TPc, 0);
  k_attn_cross<<<gblk((size_t)BNc * Hc), 256, 0, stream>>>(c1, c2, c3, c4);
  f2h(c4, hA, (size_t)BNc * Dc);
  gemm(hA, cWo, Dc, Dc, Dc, cbo, c5, BNc, 0);
  ln(c5, xm, lnc_g, lnc_b, xf, BNc);

  // ================= output heads =================
  f2h(xf, hA, (size_t)BNc * Dc);
  gemm(hA, Wmu, Dc, PREDc * Kmix, Dc, bmu, muR, BNc, 0);
  gemm(hA, Wsg, Dc, PREDc * Kmix, Dc, bsg, sgR, BNc, 0);
  gemm(hA, Wpi, Dc, PREDc * Kmix, Dc, bpi, piR, BNc, 0);
  k_heads<<<gblk((size_t)BNc * PREDc * Kmix), 256, 0, stream>>>(
      muR, sgR, piR, (float*)d_out);
}
